// GAT_emb_46712064311584
// MI455X (gfx1250) — matrix operand order repaired and checked by reading, not compiled, not run on hardware
//
#include <hip/hip_runtime.h>
#include <hip/hip_bf16.h>
#include <math.h>

// ---------------- problem constants (from reference) ----------------
constexpr int NN   = 50000;   // nodes
constexpr int EE   = 800000;  // edges
constexpr int INF_ = 128;     // input feature dim
constexpr int HIDF = 32;
constexpr int OUTF = 64;
constexpr int H1N  = 8;       // heads in layer 1
#define SLOPE 0.2f

typedef __attribute__((ext_vector_type(16))) __bf16 v16bf;
typedef __attribute__((ext_vector_type(8)))  __bf16 v8bf;
typedef __attribute__((ext_vector_type(8)))  float  v8f;

// ---------------- WMMA GEMM: C[M,N] = A[M,K] * W[K,N]  (fp32 -> bf16 -> f32 accum)
// Block = 256 threads = 8 waves. All 8 waves share one 16-column slice of W
// (tn), staged once into LDS as bf16 in fragment-contiguous layout; each wave
// computes one 16x16 tile (8 consecutive m-tiles per block). K templated so
// the k-loop fully unrolls into a v_wmma chain.
template <int K>
__global__ void gemm_bf16_wmma(const float* __restrict__ A,
                               const float* __restrict__ W,
                               float* __restrict__ C,
                               int M, int N) {
  constexpr int KQ = K / 16;               // 16-deep K blocks
  // Bl[q][col][j]: j = K index within block q, contiguous -> 2x ds_load_b128
  // per fragment. Inner dim padded to 24 (48B row) for bank spread, 16B aligned.
  __shared__ __bf16 Bl[KQ][16][24];

  const int tiles_n = N >> 4;
  const int tmg = blockIdx.x / tiles_n;    // group of 8 m-tiles
  const int tn  = blockIdx.x - tmg * tiles_n;

  // ---- stage W[:, tn*16 .. tn*16+15] into LDS (bf16) ----
  {
    const int col = threadIdx.x & 15;
    const float* Wc = W + (size_t)(tn * 16 + col);
    for (int k = (int)(threadIdx.x >> 4); k < K; k += 16) {
      Bl[k >> 4][col][k & 15] = (__bf16)Wc[(size_t)k * N];
    }
  }
  __syncthreads();

  const int wave = threadIdx.x >> 5;
  const int lane = threadIdx.x & 31;
  const int tm = tmg * 8 + wave;
  if (tm * 16 >= M) return;                // wave-uniform, EXEC all-ones for WMMA

  const int r    = lane & 15;              // row (A) / col (B,C) within tile
  const int half = lane >> 4;              // lane-half selects K-subblock

  const float* __restrict__ Arow = A + (size_t)(tm * 16 + r) * K;

  v8f acc = {};
#pragma unroll
  for (int k0 = 0; k0 < K; k0 += 32) {
    if (k0 + 32 < K) __builtin_prefetch(Arow + k0 + 32, 0, 3);  // near-cache prefetch
    // A 16x32 bf16 fragment: lanes 0-15 K=[0..15],
    // lanes 16-31 K=[16..31]  (contiguous 8-float runs -> b128 loads)
    v16bf a;
#pragma unroll
    for (int i = 0; i < 8; ++i) {
      a[i]     = (__bf16)Arow[k0 + half * 16 + i];
      a[8 + i] = (__bf16)Arow[k0 + half * 16 + 8 + i];
    }
    // B 32x16 bf16 fragment from LDS: element j -> K = (k0 + half*16) + j
    const v8bf* bp = (const v8bf*)&Bl[(k0 >> 4) + half][r][0];
    v8bf blo = bp[0], bhi = bp[1];
    v16bf b = __builtin_shufflevector(blo, bhi,
                                      0, 1, 2, 3, 4, 5, 6, 7,
                                      8, 9, 10, 11, 12, 13, 14, 15);
    acc = __builtin_amdgcn_wmma_f32_16x16x32_bf16(
        /*neg_a=*/false, a, /*neg_b=*/false, b,
        /*c_mod=*/(short)0, acc, /*reuse_a=*/false, /*reuse_b=*/false);
  }
  // C/D layout: VGPR i holds row (i + 8*half), column = lane&15
#pragma unroll
  for (int i = 0; i < 8; ++i) {
    C[(size_t)(tm * 16 + i + 8 * half) * N + tn * 16 + r] = acc[i];
  }
}

// ---------------- el/er = <h[n,h,:], al[h,:]> , <h[n,h,:], ar[h,:]>
__global__ void attn_scores(const float* __restrict__ h,
                            const float* __restrict__ al,
                            const float* __restrict__ ar,
                            float* __restrict__ el, float* __restrict__ er,
                            int nNodes, int heads, int F) {
  int idx = blockIdx.x * blockDim.x + threadIdx.x;
  if (idx >= nNodes * heads) return;
  int n  = idx / heads;
  int hd = idx - n * heads;
  const float* hp  = h + (size_t)n * heads * F + (size_t)hd * F;
  const float* alp = al + (size_t)hd * F;
  const float* arp = ar + (size_t)hd * F;
  float sl = 0.f, sr = 0.f;
  for (int f = 0; f < F; ++f) {
    float v = hp[f];
    sl += v * alp[f];
    sr += v * arp[f];
  }
  el[idx] = sl;
  er[idx] = sr;
}

// ---------------- fill
__global__ void fill_f32(float* __restrict__ p, float v, int n) {
  int idx = blockIdx.x * blockDim.x + threadIdx.x;
  if (idx < n) p[idx] = v;
}

// ---------------- edge phase 1: leaky-relu logit + segment max (monotone int-atomic trick)
__global__ void edge_logit_max(const int* __restrict__ src, const int* __restrict__ dst,
                               const float* __restrict__ el, const float* __restrict__ er,
                               float* __restrict__ logit, float* __restrict__ m,
                               int nE, int heads) {
  int idx = blockIdx.x * blockDim.x + threadIdx.x;
  if (idx >= nE * heads) return;
  int e  = idx / heads;
  int hd = idx - e * heads;
  int s_ = src[e], d_ = dst[e];
  float x = el[s_ * heads + hd] + er[d_ * heads + hd];
  x = (x > 0.f) ? x : SLOPE * x;
  logit[idx] = x;
  int* mp = (int*)(m + (size_t)d_ * heads + hd);
  if (x >= 0.f) atomicMax(mp, __float_as_int(x));
  else          atomicMin((unsigned int*)mp, __float_as_uint(x));
}

// ---------------- edge phase 2: ee = exp(e - m[dst]); s[dst] += ee  (in-place logit -> ee)
__global__ void edge_exp_sum(const int* __restrict__ dst,
                             float* __restrict__ logit, const float* __restrict__ m,
                             float* __restrict__ ssum, int nE, int heads) {
  int idx = blockIdx.x * blockDim.x + threadIdx.x;
  if (idx >= nE * heads) return;
  int e  = idx / heads;
  int hd = idx - e * heads;
  int d_ = dst[e];
  float ee = __expf(logit[idx] - m[(size_t)d_ * heads + hd]);
  logit[idx] = ee;
  atomicAdd(ssum + (size_t)d_ * heads + hd, ee);
}

// ---------------- edge phase 3: agg[dst] += h[src] * (ee / s[dst])
// One thread per (edge, head, feature): lane-contiguous features => coalesced
// h[src] gathers and coalesced global_atomic_add_f32 on agg[dst].
__global__ void edge_aggregate(const int* __restrict__ src, const int* __restrict__ dst,
                               const float* __restrict__ h,
                               const float* __restrict__ ee, const float* __restrict__ ssum,
                               float* __restrict__ agg, int nE, int heads, int F) {
  int idx = blockIdx.x * blockDim.x + threadIdx.x;
  const int HF = heads * F;
  if (idx >= nE * HF) return;
  int e   = idx / HF;
  int rem = idx - e * HF;            // = hd*F + f  (contiguous across lanes)
  int hd  = rem / F;
  int s_ = src[e], d_ = dst[e];
  float alpha = ee[(size_t)e * heads + hd] / ssum[(size_t)d_ * heads + hd];
  float v = h[(size_t)s_ * HF + rem] * alpha;
  atomicAdd(agg + (size_t)d_ * HF + rem, v);
}

// ---------------- bias (+ optional relu)
__global__ void finalize(const float* __restrict__ agg, const float* __restrict__ b,
                         float* __restrict__ out, int total, int HF, int do_relu) {
  int idx = blockIdx.x * blockDim.x + threadIdx.x;
  if (idx >= total) return;
  float v = agg[idx] + b[idx % HF];
  if (do_relu) v = v > 0.f ? v : 0.f;
  out[idx] = v;
}

static inline int cdiv(int a, int b) { return (a + b - 1) / b; }

extern "C" void kernel_launch(void* const* d_in, const int* in_sizes, int n_in,
                              void* d_out, int out_size, void* d_ws, size_t ws_size,
                              hipStream_t stream) {
  (void)in_sizes; (void)n_in; (void)out_size; (void)ws_size;
  const float* feat = (const float*)d_in[0];
  const int*   src  = (const int*)  d_in[1];
  const int*   dst  = (const int*)  d_in[2];
  const float* W1   = (const float*)d_in[3];
  const float* al1  = (const float*)d_in[4];
  const float* ar1  = (const float*)d_in[5];
  const float* b1   = (const float*)d_in[6];
  const float* W2   = (const float*)d_in[7];
  const float* al2  = (const float*)d_in[8];
  const float* ar2  = (const float*)d_in[9];
  const float* b2   = (const float*)d_in[10];
  const float* W3   = (const float*)d_in[11];
  const float* al3  = (const float*)d_in[12];
  const float* ar3  = (const float*)d_in[13];
  const float* b3   = (const float*)d_in[14];
  float* out = (float*)d_out;
  float* emb_out = out + (size_t)NN * OUTF;   // second tuple element

  // workspace layout (floats)
  float* ws = (float*)d_ws;
  float* R0 = ws;                              // NN*256  (h per layer)
  float* R1 = R0 + (size_t)NN * 256;           // NN*256  (agg / relu'd x2)
  float* R2 = R1 + (size_t)NN * 256;           // EE*8    (logits -> ee)
  float* elb = R2 + (size_t)EE * H1N;          // NN*8
  float* erb = elb + (size_t)NN * H1N;         // NN*8
  float* mb  = erb + (size_t)NN * H1N;         // NN*8
  float* sb  = mb  + (size_t)NN * H1N;         // NN*8

  const float NEG_INF = -__builtin_huge_valf();
  const int TB = 256;
  const int MGRP = cdiv(NN / 16, 8);           // 8 m-tiles per block

  // =================== Layer 1: IN=128 -> 8 heads x 32 ===================
  {
    const int heads = H1N, F = HIDF, HF = heads * F;   // HF = 256
    gemm_bf16_wmma<INF_><<<MGRP * (HF / 16), TB, 0, stream>>>(feat, W1, R0, NN, HF);
    attn_scores<<<cdiv(NN * heads, TB), TB, 0, stream>>>(R0, al1, ar1, elb, erb, NN, heads, F);
    fill_f32<<<cdiv(NN * heads, TB), TB, 0, stream>>>(mb, NEG_INF, NN * heads);
    fill_f32<<<cdiv(NN * heads, TB), TB, 0, stream>>>(sb, 0.f, NN * heads);
    fill_f32<<<cdiv(NN * HF, TB), TB, 0, stream>>>(R1, 0.f, NN * HF);
    edge_logit_max<<<cdiv(EE * heads, TB), TB, 0, stream>>>(src, dst, elb, erb, R2, mb, EE, heads);
    edge_exp_sum<<<cdiv(EE * heads, TB), TB, 0, stream>>>(dst, R2, mb, sb, EE, heads);
    edge_aggregate<<<cdiv(EE * HF, TB), TB, 0, stream>>>(src, dst, R0, R2, sb, R1, EE, heads, F);
    finalize<<<cdiv(NN * HF, TB), TB, 0, stream>>>(R1, b1, R1, NN * HF, HF, /*relu=*/1); // x2 in R1
  }

  // =================== Layer 2: 256 -> 1 head x 32 (embedding) ===================
  {
    const int heads = 1, F = HIDF, HF = F;             // 32
    gemm_bf16_wmma<H1N * HIDF><<<MGRP * (HF / 16), TB, 0, stream>>>(R1, W2, R0, NN, HF);
    attn_scores<<<cdiv(NN * heads, TB), TB, 0, stream>>>(R0, al2, ar2, elb, erb, NN, heads, F);
    fill_f32<<<cdiv(NN * heads, TB), TB, 0, stream>>>(mb, NEG_INF, NN * heads);
    fill_f32<<<cdiv(NN * heads, TB), TB, 0, stream>>>(sb, 0.f, NN * heads);
    fill_f32<<<cdiv(NN * HF, TB), TB, 0, stream>>>(R1, 0.f, NN * HF);
    edge_logit_max<<<cdiv(EE * heads, TB), TB, 0, stream>>>(src, dst, elb, erb, R2, mb, EE, heads);
    edge_exp_sum<<<cdiv(EE * heads, TB), TB, 0, stream>>>(dst, R2, mb, sb, EE, heads);
    edge_aggregate<<<cdiv(EE * HF, TB), TB, 0, stream>>>(src, dst, R0, R2, sb, R1, EE, heads, F);
    finalize<<<cdiv(NN * HF, TB), TB, 0, stream>>>(R1, b2, emb_out, NN * HF, HF, /*relu=*/0);
  }

  // =================== Layer 3: 32 -> 1 head x 64 ===================
  {
    const int heads = 1, F = OUTF, HF = F;             // 64
    gemm_bf16_wmma<HIDF><<<MGRP * (HF / 16), TB, 0, stream>>>(emb_out, W3, R0, NN, HF);
    attn_scores<<<cdiv(NN * heads, TB), TB, 0, stream>>>(R0, al3, ar3, elb, erb, NN, heads, F);
    fill_f32<<<cdiv(NN * heads, TB), TB, 0, stream>>>(mb, NEG_INF, NN * heads);
    fill_f32<<<cdiv(NN * heads, TB), TB, 0, stream>>>(sb, 0.f, NN * heads);
    fill_f32<<<cdiv(NN * HF, TB), TB, 0, stream>>>(R1, 0.f, NN * HF);
    edge_logit_max<<<cdiv(EE * heads, TB), TB, 0, stream>>>(src, dst, elb, erb, R2, mb, EE, heads);
    edge_exp_sum<<<cdiv(EE * heads, TB), TB, 0, stream>>>(dst, R2, mb, sb, EE, heads);
    edge_aggregate<<<cdiv(EE * HF, TB), TB, 0, stream>>>(src, dst, R0, R2, sb, R1, EE, heads, F);
    finalize<<<cdiv(NN * HF, TB), TB, 0, stream>>>(R1, b3, out, NN * HF, HF, /*relu=*/0);
  }
}
